// GraphSAGEModel_13305808683556
// MI455X (gfx1250) — matrix operand order repaired and checked by reading, not compiled, not run on hardware
//
#include <hip/hip_runtime.h>

typedef __attribute__((ext_vector_type(16))) _Float16 v16h;
typedef __attribute__((ext_vector_type(8)))  _Float16 v8h;
typedef __attribute__((ext_vector_type(4)))  _Float16 v4h;
typedef __attribute__((ext_vector_type(8)))  float    v8f;

#define N_NODES 100000
#define N_EDGES 1600000
#define IN_C    19
#define HID_C   128
#define OUT_C   4

// Workspace layout (float offsets). Zeroed region = [0, WS_H).
#define WS_CNT  0
#define WS_AGG1 (N_NODES)                       // N*19
#define WS_AGG2 (WS_AGG1 + N_NODES * IN_C)      // N*128
#define WS_H    (WS_AGG2 + N_NODES * HID_C)     // N*128
#define WS_TOT  (WS_H + N_NODES * HID_C)        // ~110.4 MB total

// ---------------------------------------------------------------- zero
__global__ void k_zero(float4* __restrict__ p, int n4) {
  int i = blockIdx.x * blockDim.x + threadIdx.x;
  int stride = gridDim.x * blockDim.x;
  for (; i < n4; i += stride) p[i] = make_float4(0.f, 0.f, 0.f, 0.f);
}

// ------------------------------------------------- layer-1 scatter (+degree)
__global__ __launch_bounds__(256) void k_scatter1(
    const float* __restrict__ x, const int* __restrict__ ei,
    float* __restrict__ cnt, float* __restrict__ agg1) {
  const int wave = threadIdx.x >> 5, lane = threadIdx.x & 31;
  const int e = blockIdx.x * 8 + wave;          // grid = E/8 exact
  const int s = ei[e];
  const int d = ei[N_EDGES + e];
  if (lane == 0) atomicAdd(&cnt[d], 1.0f);
  if (lane < IN_C) atomicAdd(&agg1[d * IN_C + lane], x[s * IN_C + lane]);
}

// ------------------------------------------------- layer-2 scatter (128 ch)
__global__ __launch_bounds__(256) void k_scatter2(
    const float* __restrict__ h, const int* __restrict__ ei,
    float* __restrict__ agg2) {
  const int wave = threadIdx.x >> 5, lane = threadIdx.x & 31;
  const int e = blockIdx.x * 8 + wave;          // grid = E/8 exact
  const int s = ei[e];
  const int d = ei[N_EDGES + e];
  const float* __restrict__ src = h + (size_t)s * HID_C;
  float* __restrict__ dst = agg2 + (size_t)d * HID_C;
  #pragma unroll
  for (int i = 0; i < 4; ++i)
    atomicAdd(&dst[i * 32 + lane], src[i * 32 + lane]);
}

// ------------------------------------------------- layer-1 WMMA GEMM + relu
// block: 256 thr = 8 waves; 16 nodes x 128 cols; A=[mean|x] K 38->64
__global__ __launch_bounds__(256) void k_gemm1(
    const float* __restrict__ x, const float* __restrict__ agg1,
    const float* __restrict__ cnt, const float* __restrict__ w_l,
    const float* __restrict__ w_r, const float* __restrict__ bias,
    float* __restrict__ h) {
  __shared__ __align__(32) _Float16 As[16 * 64];    // [row][k] 2 KB
  __shared__ __align__(32) _Float16 Bs[128 * 64];   // [col][k] 16 KB

  const int tid  = threadIdx.x;
  const int wave = tid >> 5, lane = tid & 31;
  const int base = blockIdx.x * 16;                 // grid = N/16 exact

  // fill Bs (coalesced over c; k wave-uniform), zero-padded K
  #pragma unroll
  for (int it = 0; it < 32; ++it) {
    const int i = tid + it * 256;                   // 8192 elements
    const int c = i & 127, k = i >> 7;
    float v = 0.f;
    if (k < IN_C)          v = w_l[k * HID_C + c];
    else if (k < 2 * IN_C) v = w_r[(k - IN_C) * HID_C + c];
    Bs[c * 64 + k] = (_Float16)v;
  }
  // fill As (zero-padded), mean = agg1/cnt
  #pragma unroll
  for (int it = 0; it < 4; ++it) {
    const int i = tid + it * 256;                   // 1024 elements
    const int row = i >> 6, k = i & 63;
    const int node = base + row;
    float v = 0.f;
    if (k < IN_C)          v = agg1[node * IN_C + k] / fmaxf(cnt[node], 1.0f);
    else if (k < 2 * IN_C) v = x[node * IN_C + (k - IN_C)];
    As[i] = (_Float16)v;
  }
  __syncthreads();

  const int row = lane & 15;
  const int kbA = (lane < 16) ? 0 : 16;
  const int kbB = (lane < 16) ? 0 : 16;
  const int cg  = wave * 16 + (lane & 15);

  v8f acc = {};
  #pragma unroll
  for (int kk = 0; kk < 2; ++kk) {
    const int kk0 = kk * 32;
    const v8h alo = *(const v8h*)(As + row * 64 + kk0 + kbA);
    const v8h ahi = *(const v8h*)(As + row * 64 + kk0 + kbA + 8);
    const v16h a = __builtin_shufflevector(alo, ahi,
        0, 1, 2, 3, 4, 5, 6, 7, 8, 9, 10, 11, 12, 13, 14, 15);
    const v16h b = *(const v16h*)(Bs + cg * 64 + kk0 + kbB);
    acc = __builtin_amdgcn_wmma_f32_16x16x32_f16(
        false, a, false, b, (short)0, acc, false, false);
  }

  const int mb = (lane < 16) ? 0 : 8;
  const float bv = bias[cg];
  #pragma unroll
  for (int j = 0; j < 8; ++j) {
    float v = fmaxf(acc[j] + bv, 0.0f);             // relu
    h[(size_t)(base + mb + j) * HID_C + cg] = v;
  }
}

// --------------------------------- layer-2 WMMA GEMM + bias + log_softmax
// block: 320 thr = 10 waves; each wave = one 16-node tile; K=256=[mean2|h]
__global__ __launch_bounds__(320) void k_gemm2(
    const float* __restrict__ h, const float* __restrict__ agg2,
    const float* __restrict__ cnt, const float* __restrict__ w_l,
    const float* __restrict__ w_r, const float* __restrict__ bias,
    float* __restrict__ out) {
  __shared__ __align__(32) _Float16 Bs2[16 * 256];        // [col][k] 8 KB
  __shared__ __align__(32) _Float16 As2[10 * 16 * 256];   // per-wave 8 KB

  const int tid  = threadIdx.x;
  const int wave = tid >> 5, lane = tid & 31;
  const int tile = blockIdx.x * 10 + wave;          // grid = N/160 exact
  const int base = tile * 16;

  // fill Bs2 (zero-padded N 4->16)
  for (int i = tid; i < 16 * 256; i += 320) {
    const int c = i >> 8, k = i & 255;
    float v = 0.f;
    if (c < OUT_C)
      v = (k < HID_C) ? w_l[k * OUT_C + c] : w_r[(k - HID_C) * OUT_C + c];
    Bs2[c * 256 + k] = (_Float16)v;
  }

  // fill this wave's As2 slab, branch-free per lane:
  // lane l -> row l&15; lanes 0..15 write K[0,128)=mean2, 16..31 write K[128,256)=h
  {
    _Float16* slab = As2 + wave * (16 * 256);
    const int row  = lane & 15;
    const int node = base + row;
    const bool selfHalf = (lane >= 16);
    const float scale = selfHalf ? 1.0f : (1.0f / fmaxf(cnt[node], 1.0f));
    const float4* __restrict__ src = (const float4*)(
        (selfHalf ? h : agg2) + (size_t)node * HID_C);
    _Float16* dst = slab + row * 256 + (selfHalf ? 128 : 0);
    #pragma unroll
    for (int j = 0; j < 32; ++j) {                  // 128 values, float4 loads
      const float4 v = src[j];
      v4h p;
      p[0] = (_Float16)(v.x * scale);
      p[1] = (_Float16)(v.y * scale);
      p[2] = (_Float16)(v.z * scale);
      p[3] = (_Float16)(v.w * scale);
      *(v4h*)(dst + 4 * j) = p;
    }
  }
  __syncthreads();

  const _Float16* slab = As2 + wave * (16 * 256);
  const int row = lane & 15;
  const int kbA = (lane < 16) ? 0 : 16;
  const int kbB = (lane < 16) ? 0 : 16;
  const int cg  = lane & 15;

  v8f acc = {};
  #pragma unroll
  for (int kk = 0; kk < 8; ++kk) {
    const int kk0 = kk * 32;
    const v8h alo = *(const v8h*)(slab + row * 256 + kk0 + kbA);
    const v8h ahi = *(const v8h*)(slab + row * 256 + kk0 + kbA + 8);
    const v16h a = __builtin_shufflevector(alo, ahi,
        0, 1, 2, 3, 4, 5, 6, 7, 8, 9, 10, 11, 12, 13, 14, 15);
    const v16h b = *(const v16h*)(Bs2 + cg * 256 + kk0 + kbB);
    acc = __builtin_amdgcn_wmma_f32_16x16x32_f16(
        false, a, false, b, (short)0, acc, false, false);
  }

  // log_softmax over the 4 valid columns (lane quad holds one node's logits)
  const int mb = (lane < 16) ? 0 : 8;
  const float bv = (cg < OUT_C) ? bias[cg] : 0.0f;
  #pragma unroll
  for (int j = 0; j < 8; ++j) {
    float v = acc[j] + bv;
    float m = fmaxf(v, __shfl_xor(v, 1));
    m = fmaxf(m, __shfl_xor(m, 2));
    float e = __expf(v - m);
    float s = e + __shfl_xor(e, 1);
    s += __shfl_xor(s, 2);
    float r = v - m - __logf(s);
    if (cg < OUT_C) out[(size_t)(base + mb + j) * OUT_C + cg] = r;
  }
}

// ---------------------------------------------------------------- launch
extern "C" void kernel_launch(void* const* d_in, const int* in_sizes, int n_in,
                              void* d_out, int out_size, void* d_ws, size_t ws_size,
                              hipStream_t stream) {
  const float* x    = (const float*)d_in[0];
  const int*   ei   = (const int*)d_in[1];   // edge_index (int32 per bridge doc)
  const float* w1_l = (const float*)d_in[2];
  const float* w1_r = (const float*)d_in[3];
  const float* b1   = (const float*)d_in[4];
  const float* w2_l = (const float*)d_in[5];
  const float* w2_r = (const float*)d_in[6];
  const float* b2   = (const float*)d_in[7];
  float* out = (float*)d_out;

  float* ws   = (float*)d_ws;
  float* cnt  = ws + WS_CNT;
  float* agg1 = ws + WS_AGG1;
  float* agg2 = ws + WS_AGG2;
  float* h    = ws + WS_H;

  // zero cnt+agg1+agg2 (contiguous prefix, WS_H floats, divisible by 4)
  k_zero<<<4096, 256, 0, stream>>>((float4*)ws, WS_H / 4);
  k_scatter1<<<N_EDGES / 8, 256, 0, stream>>>(x, ei, cnt, agg1);
  k_gemm1<<<N_NODES / 16, 256, 0, stream>>>(x, agg1, cnt, w1_l, w1_r, b1, h);
  k_scatter2<<<N_EDGES / 8, 256, 0, stream>>>(h, ei, agg2);
  k_gemm2<<<N_NODES / 160, 320, 0, stream>>>(h, agg2, cnt, w2_l, w2_r, b2, out);
}
